// OuterProductMean_9208409882905
// MI455X (gfx1250) — matrix operand order repaired and checked by reading, not compiled, not run on hardware
//
#include <hip/hip_runtime.h>

#define NSEQ 512
#define NRES 256
#define CM   256
#define CO   32
#define CZ   128

typedef __attribute__((ext_vector_type(16))) __bf16 v16bf;
typedef __attribute__((ext_vector_type(8)))  float  v8f;
typedef __attribute__((ext_vector_type(4)))  unsigned int u4;
typedef __attribute__((ext_vector_type(4)))  int v4i;

union Frag { v16bf v; u4 q[2]; };

static __device__ __forceinline__ v8f wmma_bf16(Frag a, Frag b, v8f c) {
  return __builtin_amdgcn_wmma_f32_16x16x32_bf16(false, a.v, false, b.v,
                                                 (short)0, c, false, false);
}

// Async global->LDS path (CDNA5 ASYNCcnt engine), guarded so host pass and
// older toolchains fall back to reg-staged copies.
// Probe-derived signature: (v4i addrspace(1)* gsrc, v4i addrspace(3)* ldst,
//                           imm int offset, imm int cpol)
#if __has_builtin(__builtin_amdgcn_global_load_async_to_lds_b128) && \
    __has_builtin(__builtin_amdgcn_s_wait_asynccnt)
#define USE_ASYNC_LDS 1
typedef __attribute__((address_space(1))) v4i GVec;
typedef __attribute__((address_space(3))) v4i LVec;
#else
#define USE_ASYNC_LDS 0
#endif

// ---------------------------------------------------------------------------
// K0: pack weights. wt_cat[c(64)][m(256)] = [left_w | right_w]^T (bf16)
//     wt_out[f(128)][ce(1024)] = output_w^T (bf16); bias_cat = [lb | rb]
// ---------------------------------------------------------------------------
__global__ void prep_weights(const float* __restrict__ lw, const float* __restrict__ rw,
                             const float* __restrict__ lb, const float* __restrict__ rb,
                             const float* __restrict__ ow,
                             __bf16* __restrict__ wt_cat, float* __restrict__ bias_cat,
                             __bf16* __restrict__ wt_out) {
  const int t = blockIdx.x * blockDim.x + threadIdx.x;
  const int nt = gridDim.x * blockDim.x;
  for (int i = t; i < 64 * 256; i += nt) {
    const int c = i >> 8, m = i & 255;
    const float v = (c < 32) ? lw[m * 32 + c] : rw[m * 32 + (c - 32)];
    wt_cat[i] = (__bf16)v;
  }
  for (int i = t; i < 64; i += nt) bias_cat[i] = (i < 32) ? lb[i] : rb[i - 32];
  for (int i = t; i < 128 * 1024; i += nt) {
    const int f = i >> 10, ce = i & 1023;
    wt_out[i] = (__bf16)ow[ce * 128 + f];
  }
}

// ---------------------------------------------------------------------------
// K1: norm[b][d] = sum_a mask[a][b]*mask[a][d]
// ---------------------------------------------------------------------------
__global__ void mask_norm(const float* __restrict__ mask, float* __restrict__ norm) {
  const int b = blockIdx.x, d = threadIdx.x;
  float s = 0.f;
  for (int a = 0; a < NSEQ; ++a)
    s += mask[a * NRES + b] * mask[a * NRES + d];
  norm[b * NRES + d] = s;
}

// ---------------------------------------------------------------------------
// K2: LayerNorm + left/right projections via WMMA.
//     Block = 256 threads = 8 waves, covers 32 'a' rows for one 'b'.
//     Output stored K-major: left_t[(b*32+c)][a], right_t[(b*32+c)][a] (bf16)
// ---------------------------------------------------------------------------
__global__ void __launch_bounds__(256)
ln_project(const float* __restrict__ act, const float* __restrict__ mask,
           const float* __restrict__ ln_scale, const float* __restrict__ ln_offset,
           const __bf16* __restrict__ wt_cat, const float* __restrict__ bias_cat,
           __bf16* __restrict__ left_t, __bf16* __restrict__ right_t) {
  __shared__ __bf16 xs[32][CM];   // 16 KB LN'd activations (bf16)

  const int tid  = threadIdx.x;
  const int lane = tid & 31;
  const int w    = tid >> 5;
  const int lh   = lane & 15;
  const bool hi  = lane >= 16;
  const int a0   = blockIdx.x * 32;
  const int b    = blockIdx.y;

  // Phase 1: LayerNorm 4 rows per wave (32 lanes cooperate per row)
  #pragma unroll
  for (int rr = 0; rr < 4; ++rr) {
    const int ar = w * 4 + rr;
    const float* src = act + ((size_t)(a0 + ar) * NRES + b) * CM;
    float x[8], s = 0.f, s2 = 0.f;
    #pragma unroll
    for (int j = 0; j < 8; ++j) {
      x[j] = src[lane + 32 * j];
      s += x[j]; s2 += x[j] * x[j];
    }
    #pragma unroll
    for (int o = 16; o >= 1; o >>= 1) {
      s  += __shfl_xor(s,  o, 32);
      s2 += __shfl_xor(s2, o, 32);
    }
    const float mu = s * (1.f / CM);
    const float rs = rsqrtf(s2 * (1.f / CM) - mu * mu + 1e-5f);
    #pragma unroll
    for (int j = 0; j < 8; ++j) {
      const int m = lane + 32 * j;
      xs[ar][m] = (__bf16)((x[j] - mu) * rs * ln_scale[m] + ln_offset[m]);
    }
  }
  __syncthreads();

  // Phase 2: [32 a] x [256 m] x [64 c] GEMM. Wave (mt=w>>2, nt=w&3).
  const int mbase = (w >> 2) * 16;
  const int nbase = (w & 3) * 16;
  v8f acc = {0.f, 0.f, 0.f, 0.f, 0.f, 0.f, 0.f, 0.f};
  #pragma unroll
  for (int k0 = 0; k0 < CM; k0 += 32) {
    Frag af, bf_;
    const int row = mbase + lh;
    const int ka  = k0 + (hi ? 8 : 0);
    af.q[0] = ((const u4*)&xs[row][ka])[0];
    af.q[1] = ((const u4*)&xs[row][ka + 16])[0];
    const int c  = nbase + lh;
    const int kb = k0 + (hi ? 8 : 0);
    const __bf16* bp = wt_cat + c * CM + kb;
    bf_.q[0] = ((const u4*)bp)[0];
    bf_.q[1] = ((const u4*)(bp + 16))[0];
    acc = wmma_bf16(af, bf_, acc);
  }

  // Phase 3: apply bias+mask, store transposed (K-major) as bf16
  const int c = nbase + lh;
  const float bias = bias_cat[c];
  #pragma unroll
  for (int r = 0; r < 8; ++r) {
    const int arow = mbase + r + (hi ? 8 : 0);
    const int a = a0 + arow;
    const float mv = mask[(size_t)a * NRES + b];
    const __bf16 bv = (__bf16)((acc[r] + bias) * mv);
    if (c < 32) left_t [((size_t)(b * 32 + c)) * NSEQ + a]        = bv;
    else        right_t[((size_t)(b * 32 + (c - 32))) * NSEQ + a] = bv;
  }
}

// ---------------------------------------------------------------------------
// K3: fused outer-product GEMM + W contraction.
//     Grid (64,64): workgroup -> 4 b's x 4 d's = 128x128 (M,N) op tile, K=512.
//     Stage 1: bf16 WMMA main loop; tiles land in LDS either via the CDNA5
//              async global->LDS engine (ASYNCcnt + s_wait_asynccnt) or via
//              reg staging fallback. Double-buffered 32 KB LDS.
//     Stage 2: op tile -> bf16 [16 pairs][1024 ce] in LDS (aliases stage),
//              16x128x1024 WMMA against Wt[f][ce], bias + 1/(eps+norm) store.
// ---------------------------------------------------------------------------
__global__ void __launch_bounds__(256)
outer_main(const __bf16* __restrict__ left_t, const __bf16* __restrict__ right_t,
           const __bf16* __restrict__ wt_out, const float* __restrict__ out_bias,
           const float* __restrict__ norm, float* __restrict__ out) {
  __shared__ alignas(16) char smem[32 * 1024];
  __bf16* stL  = (__bf16*)smem;                // [2][128][32]
  __bf16* stR  = (__bf16*)(smem + 16 * 1024);  // [2][128][32]
  __bf16* opbf = (__bf16*)smem;                // [16][1024], aliases stage afterwards

  const int tid  = threadIdx.x;
  const int lane = tid & 31;
  const int w    = tid >> 5;
  const int lh   = lane & 15;
  const bool hi  = lane >= 16;

  const int b0 = blockIdx.x * 4;
  const int d0 = blockIdx.y * 4;
  const size_t m0 = (size_t)b0 * 32;
  const size_t n0 = (size_t)d0 * 32;

  const v8f vzero = {0.f, 0.f, 0.f, 0.f, 0.f, 0.f, 0.f, 0.f};
  v8f acc[2][4];
  #pragma unroll
  for (int i = 0; i < 2; ++i)
    #pragma unroll
    for (int j = 0; j < 4; ++j) acc[i][j] = vzero;

#if USE_ASYNC_LDS
  // Each wave issues 2 async b128 instructions per matrix per K-slab:
  // one instruction = 32 lanes x 16B = 512B = 8 rows of 64B.
  const int  crow = w * 16 + (lane >> 2);        // rows w*16 .. w*16+7 (j=0)
  const int  cbyt = (lane & 3) * 16;
  const char* gl0 = (const char*)left_t  + (m0 + crow) * (NSEQ * 2) + cbyt;
  const char* gr0 = (const char*)right_t + (n0 + crow) * (NSEQ * 2) + cbyt;
  const unsigned lofs = (unsigned)(crow * 64 + cbyt);
  char* lbaseL = smem;
  char* lbaseR = smem + 16 * 1024;

  #define ISSUE_SLAB(kk, buf)                                                         \
    do {                                                                              \
      const int  _g  = (kk) * 64;                                                     \
      const unsigned _lb = (unsigned)(buf) * 8192u;                                   \
      __builtin_amdgcn_global_load_async_to_lds_b128(                                 \
          (GVec*)(gl0 + _g),            (LVec*)(lbaseL + _lb + lofs),        0, 0);   \
      __builtin_amdgcn_global_load_async_to_lds_b128(                                 \
          (GVec*)(gl0 + _g + 8 * 1024), (LVec*)(lbaseL + _lb + lofs + 512),  0, 0);   \
      __builtin_amdgcn_global_load_async_to_lds_b128(                                 \
          (GVec*)(gr0 + _g),            (LVec*)(lbaseR + _lb + lofs),        0, 0);   \
      __builtin_amdgcn_global_load_async_to_lds_b128(                                 \
          (GVec*)(gr0 + _g + 8 * 1024), (LVec*)(lbaseR + _lb + lofs + 512),  0, 0);   \
    } while (0)

  ISSUE_SLAB(0, 0);
#else
  const int grow = tid >> 1;          // 0..127 (2 threads per row)
  const int gcol = (tid & 1) * 16;    // 16 bf16 each
  const __bf16* Lbase = left_t  + (m0 + grow) * NSEQ + gcol;
  const __bf16* Rbase = right_t + (n0 + grow) * NSEQ + gcol;

  u4 pl0, pl1, pr0, pr1;
  auto g_load = [&](int kk) {
    const u4* l = (const u4*)(Lbase + kk * 32);
    const u4* r = (const u4*)(Rbase + kk * 32);
    pl0 = l[0]; pl1 = l[1]; pr0 = r[0]; pr1 = r[1];
  };
  auto s_store = [&](int buf) {
    u4* dl = (u4*)(stL + buf * 4096 + grow * 32 + gcol);
    dl[0] = pl0; dl[1] = pl1;
    u4* dr = (u4*)(stR + buf * 4096 + grow * 32 + gcol);
    dr[0] = pr0; dr[1] = pr1;
  };

  g_load(0);
  s_store(0);
  __syncthreads();
#endif

  for (int kk = 0; kk < 16; ++kk) {
    const int buf = kk & 1;

#if USE_ASYNC_LDS
    if (kk < 15) {
      ISSUE_SLAB(kk + 1, buf ^ 1);                 // prefetch next slab into other buffer
      __builtin_amdgcn_s_wait_asynccnt(4);         // this slab's 4 instrs complete
    } else {
      __builtin_amdgcn_s_wait_asynccnt(0);
    }
    __syncthreads();                               // all waves' slab data visible
#else
    if (kk + 1 < 16) g_load(kk + 1);               // prefetch next K slab into regs
#endif

    Frag afr[2], bfr[4];
    #pragma unroll
    for (int im = 0; im < 2; ++im) {
      const int row = (w & 3) * 32 + im * 16 + lh;
      const int ka  = hi ? 8 : 0;
      const __bf16* p = stL + buf * 4096 + row * 32 + ka;
      afr[im].q[0] = ((const u4*)p)[0];
      afr[im].q[1] = ((const u4*)(p + 16))[0];
    }
    #pragma unroll
    for (int jn = 0; jn < 4; ++jn) {
      const int col = (w >> 2) * 64 + jn * 16 + lh;
      const int kb  = hi ? 8 : 0;
      const __bf16* p = stR + buf * 4096 + col * 32 + kb;
      bfr[jn].q[0] = ((const u4*)p)[0];
      bfr[jn].q[1] = ((const u4*)(p + 16))[0];
    }
    #pragma unroll
    for (int im = 0; im < 2; ++im)
      #pragma unroll
      for (int jn = 0; jn < 4; ++jn)
        acc[im][jn] = wmma_bf16(afr[im], bfr[jn], acc[im][jn]);

    __syncthreads();                               // done reading buf before overwrite
#if !USE_ASYNC_LDS
    if (kk + 1 < 16) {
      s_store(buf ^ 1);
      __syncthreads();
    }
#endif
  }

  // Rewrite op tile as bf16 [pair p=(bl*4+dl)][ce=c*32+e]
  #pragma unroll
  for (int im = 0; im < 2; ++im) {
    #pragma unroll
    for (int jn = 0; jn < 4; ++jn) {
      const int nloc = (w >> 2) * 64 + jn * 16 + lh;
      const int dl = nloc >> 5, e = nloc & 31;
      const int bl = (w & 3);
      #pragma unroll
      for (int r = 0; r < 8; ++r) {
        const int c = im * 16 + r + (hi ? 8 : 0);
        opbf[(bl * 4 + dl) * 1024 + c * 32 + e] = (__bf16)acc[im][jn][r];
      }
    }
  }
  __syncthreads();

  // Epilogue GEMM: [16 pairs] x [1024 ce] x [128 f]; wave w owns f-slice w*16.
  v8f acc2 = vzero;
  const __bf16* arow = opbf + lh * 1024 + (hi ? 8 : 0);
  const __bf16* wrow = wt_out + (size_t)(w * 16 + lh) * 1024 + (hi ? 8 : 0);
  for (int k0 = 0; k0 < 1024; k0 += 32) {
    Frag af, bf_;
    af.q[0]  = ((const u4*)(arow + k0))[0];
    af.q[1]  = ((const u4*)(arow + k0 + 16))[0];
    bf_.q[0] = ((const u4*)(wrow + k0))[0];
    bf_.q[1] = ((const u4*)(wrow + k0 + 16))[0];
    acc2 = wmma_bf16(af, bf_, acc2);
  }

  const int f = w * 16 + lh;
  const float bias = out_bias[f];
  #pragma unroll
  for (int r = 0; r < 8; ++r) {
    const int p = r + (hi ? 8 : 0);
    const int b = b0 + (p >> 2), d = d0 + (p & 3);
    const float nv = norm[b * NRES + d];
    out[((size_t)b * NRES + d) * CZ + f] = (acc2[r] + bias) / (1e-3f + nv);
  }
}

// ---------------------------------------------------------------------------
extern "C" void kernel_launch(void* const* d_in, const int* in_sizes, int n_in,
                              void* d_out, int out_size, void* d_ws, size_t ws_size,
                              hipStream_t stream) {
  (void)in_sizes; (void)n_in; (void)out_size; (void)ws_size;
  const float* act  = (const float*)d_in[0];
  const float* mask = (const float*)d_in[1];
  const float* ln_s = (const float*)d_in[2];
  const float* ln_o = (const float*)d_in[3];
  const float* lw   = (const float*)d_in[4];
  const float* lb   = (const float*)d_in[5];
  const float* rw   = (const float*)d_in[6];
  const float* rb   = (const float*)d_in[7];
  const float* ow   = (const float*)d_in[8];
  const float* ob   = (const float*)d_in[9];
  float* out = (float*)d_out;

  char* ws = (char*)d_ws;
  __bf16* left_t  = (__bf16*)(ws);                                  // 8 MB
  __bf16* right_t = (__bf16*)(ws + 8 * 1024 * 1024);                // 8 MB
  float*  normbd  = (float*) (ws + 16 * 1024 * 1024);               // 256 KB
  __bf16* wt_out  = (__bf16*)(ws + 16 * 1024 * 1024 + 256 * 1024);  // 256 KB
  __bf16* wt_cat  = (__bf16*)(ws + 16 * 1024 * 1024 + 512 * 1024);  // 32 KB
  float*  biascat = (float*) (ws + 16 * 1024 * 1024 + 544 * 1024);  // 256 B

  prep_weights<<<128, 256, 0, stream>>>(lw, rw, lb, rb, ow, wt_cat, biascat, wt_out);
  mask_norm<<<NRES, NRES, 0, stream>>>(mask, normbd);
  ln_project<<<dim3(NSEQ / 32, NRES), 256, 0, stream>>>(act, mask, ln_s, ln_o,
                                                        wt_cat, biascat, left_t, right_t);
  outer_main<<<dim3(NRES / 4, NRES / 4), 256, 0, stream>>>(left_t, right_t, wt_out,
                                                           ob, normbd, out);
}
